// MotionSqueeze_70076686401625
// MI455X (gfx1250) — matrix operand order repaired and checked by reading, not compiled, not run on hardware
//
#include <hip/hip_runtime.h>
#include <hip/hip_bf16.h>

// ---------------------------------------------------------------------------
// MotionSqueeze block for MI455X (gfx1250, wave32, WMMA).
//   k1 : 1x1 conv 128->64 (WMMA f16) + BN/ReLU + per-pixel L2 norm, weights
//        LDS-resident across 7 N-tiles per block -> feat f16 [BT][pix][64]
//   k2 : 15x15 correlation + soft-argmax flow; corr cached in LDS (f16),
//        single dot-product pass; analytic softmax max = temp*max(corr)
//   k3 : dw1 7x7 g3 + BN/ReLU + pw1 3->16 + BN/ReLU (frame duplication map)
//   k4 : dw2 3x3 g16 + pw2 16->32
//   k5 : dw3 3x3 g32 + pw3 32->64
//   k6a: dw4 3x3 g64 + BN/ReLU -> NHWC f16 (vectorized b128 stores)
//   k6b: pw4 64->128 (WMMA f16) + BN/ReLU + residual(x), weights LDS-resident
// ---------------------------------------------------------------------------

typedef _Float16 half_t;
typedef __attribute__((ext_vector_type(16))) _Float16 v16h;
typedef __attribute__((ext_vector_type(8)))  _Float16 v8h;
typedef __attribute__((ext_vector_type(8)))  float    v8f;

#define BT   128
#define HW   784
#define WID  28

__device__ __forceinline__ v16h ld8x2(const half_t* p0, const half_t* p1) {
  v8h lo = *(const v8h*)p0;
  v8h hi = *(const v8h*)p1;
  return __builtin_shufflevector(lo, hi, 0,1,2,3,4,5,6,7,8,9,10,11,12,13,14,15);
}
__device__ __forceinline__ v16h ld16(const half_t* p) {
  v8h lo = *(const v8h*)p;
  v8h hi = *(const v8h*)(p + 8);
  return __builtin_shufflevector(lo, hi, 0,1,2,3,4,5,6,7,8,9,10,11,12,13,14,15);
}

// ---------------------------------------------------------------------------
// k1: relu(bn(W[64,128] @ X[128,N])) then per-pixel L2 norm -> f16 NHWC.
// Block = 128 threads (4 waves), weights loaded once, 7 N-tiles of 16 pixels.
// ---------------------------------------------------------------------------
__global__ __launch_bounds__(128) void k1_reduce_norm(
    const float* __restrict__ x, const float* __restrict__ w_red,
    const float* __restrict__ s_red, const float* __restrict__ b_red,
    half_t* __restrict__ feat)
{
  __shared__ half_t ldsW[64 * 128];     // [row][k]
  __shared__ half_t ldsX[16 * 136];     // [pix][k], pad 8 halves
  __shared__ float  lds_s[64], lds_b[64];
  __shared__ float  sumsq[16];

  const int tid = threadIdx.x;
  const int bt  = blockIdx.x / 7;
  const int tg  = blockIdx.x % 7;       // 112-pixel group

  for (int i = tid; i < 64 * 128; i += 128) ldsW[i] = (half_t)w_red[i];
  if (tid < 64) { lds_s[tid] = s_red[tid]; lds_b[tid] = b_red[tid]; }

  const int wave = tid >> 5, lane = tid & 31;
  const int row = lane & 15, kh = lane >> 4;
  const int mbase = wave * 16 + kh * 8;

  for (int t = 0; t < 7; t++) {
    const int n0 = tg * 112 + t * 16;
    if (tid < 16) sumsq[tid] = 0.0f;
    for (int i = tid; i < 128 * 16; i += 128) {
      const int ch = i >> 4, px = i & 15;
      ldsX[px * 136 + ch] = (half_t)x[((size_t)bt * 128 + ch) * HW + n0 + px];
    }
    __syncthreads();

    v8f acc = {};
#pragma unroll
    for (int kb = 0; kb < 128; kb += 32) {
      v16h a = ld8x2(&ldsW[(wave * 16 + row) * 128 + kb + kh * 8],
                     &ldsW[(wave * 16 + row) * 128 + kb + 16 + kh * 8]);
      v16h bf = ld8x2(&ldsX[row * 136 + kb + kh * 8], &ldsX[row * 136 + kb + 16 + kh * 8]);
      acc = __builtin_amdgcn_wmma_f32_16x16x32_f16(false, a, false, bf,
                                                   (short)0, acc, false, false);
    }

    float vals[8];
    float sq = 0.0f;
#pragma unroll
    for (int v = 0; v < 8; v++) {
      float tv = acc[v] * lds_s[mbase + v] + lds_b[mbase + v];
      tv = tv > 0.0f ? tv : 0.0f;
      vals[v] = tv;
      sq += tv * tv;
    }
    atomicAdd(&sumsq[row], sq);          // ds_add_f32 cross-wave reduction
    __syncthreads();
    const float rn = rsqrtf(sumsq[row] + 1e-6f);

    v8h h;
#pragma unroll
    for (int v = 0; v < 8; v++) h[v] = (half_t)(vals[v] * rn);
    *(v8h*)&feat[((size_t)bt * HW + n0 + row) * 64 + mbase] = h;
    __syncthreads();                     // protect ldsX/sumsq for next tile
  }
}

// ---------------------------------------------------------------------------
// k2: correlation + soft-argmax flow. One thread per (pair, pixel).
// Single dot-product pass; 225 corr values cached as f16 in LDS.
// Softmax max is exactly temperature * max(corr) (gauss==1 at argmax).
// ---------------------------------------------------------------------------
#define K2T 128
__global__ __launch_bounds__(K2T) void k2_corr_flow(
    const half_t* __restrict__ feat, const int* __restrict__ temp_p,
    float* __restrict__ fbuf)
{
  __shared__ half_t cc[225 * K2T];      // [disp][tid], 57.6 KB

  const int tid = threadIdx.x;
  const int gid = blockIdx.x * K2T + tid;   // 686*128 == 112*784 exactly
  const int b = gid / HW, s = gid % HW;
  const int y = s / WID, x0 = s % WID;
  const int g = b / 7, t = b % 7;
  const int fr1 = g * 8 + t, fr2 = fr1 + 1;

  const half_t* p1 = feat + ((size_t)fr1 * HW + s) * 64;
  float f1v[64];
#pragma unroll
  for (int i = 0; i < 64; i++) f1v[i] = (float)p1[i];
  const float temp = (float)temp_p[0];

  // pass 1: compute all correlations once, cache f16, track f32 max/argmax
  float maxc = -1.0f;
  int iy = 0, ix = 0;
  int d = 0;
  for (int dy = 0; dy < 15; dy++) {
    const int yy = y + dy - 7;
    for (int dx = 0; dx < 15; dx++) {
      const int xx = x0 + dx - 7;
      float c = 0.0f;
      if (yy >= 0 && yy < WID && xx >= 0 && xx < WID) {
        const half_t* p2 = feat + ((size_t)fr2 * HW + yy * WID + xx) * 64;
#pragma unroll
        for (int i = 0; i < 64; i++) c += f1v[i] * (float)p2[i];
      }
      c = c > 0.0f ? c : 0.0f;
      cc[d * K2T + tid] = (half_t)c;
      if (c > maxc) { maxc = c; iy = dy; ix = dx; }
      d++;
    }
  }

  // pass 2: LDS-only softmax / soft-argmax sweep
  const float mmax = temp * maxc;
  float sum = 0.0f, fx = 0.0f, fy = 0.0f;
  d = 0;
  for (int dy = 0; dy < 15; dy++) {
    const float dyf = (float)(dy - iy);
    for (int dx = 0; dx < 15; dx++) {
      const float c = (float)cc[d * K2T + tid];
      const float dxf = (float)(dx - ix);
      const float gauss = __expf(-(dxf * dxf + dyf * dyf) * 0.02f); // 1/(2*5^2)
      const float e = __expf(gauss * c * temp - mmax);
      sum += e;
      fx += e * (float)(dx - 7);
      fy += e * (float)(dy - 7);
      d++;
    }
  }
  const float inv = 1.0f / (sum * 7.0f); // / DISP
  float* fb = fbuf + (size_t)b * 3 * HW;
  fb[s]          = fx * inv;
  fb[HW + s]     = fy * inv;
  fb[2 * HW + s] = maxc;
}

// ---------------------------------------------------------------------------
// k3: dw1 7x7 g3 pad3 + BN/ReLU + pw1 3->16 + BN/ReLU, with frame-dup map.
// ---------------------------------------------------------------------------
__global__ __launch_bounds__(256) void k3_dw1_pw1(
    const float* __restrict__ fbuf,
    const float* __restrict__ dw1, const float* __restrict__ pw1,
    const float* __restrict__ s1a, const float* __restrict__ b1a,
    const float* __restrict__ s1b, const float* __restrict__ b1b,
    half_t* __restrict__ t1)
{
  __shared__ float w[233];
  const int tid = threadIdx.x;
  if (tid < 147)       w[tid] = dw1[tid];
  else if (tid < 195)  w[tid] = pw1[tid - 147];
  else if (tid < 198)  w[tid] = s1a[tid - 195];
  else if (tid < 201)  w[tid] = b1a[tid - 198];
  else if (tid < 217)  w[tid] = s1b[tid - 201];
  else if (tid < 233)  w[tid] = b1b[tid - 217];
  __syncthreads();

  const int gid = blockIdx.x * 256 + tid;
  if (gid >= BT * HW) return;
  const int ft = gid / HW, s = gid % HW;
  const int y = s / WID, x = s % WID;
  const int tt = ft & 7;
  const int p = (ft >> 3) * 7 + (tt < 7 ? tt : 6);
  const float* fb = fbuf + (size_t)p * 3 * HW;

  float d[3];
#pragma unroll
  for (int c = 0; c < 3; c++) {
    float a = 0.0f;
    for (int ky = 0; ky < 7; ky++) {
      const int yy = y + ky - 3;
      if (yy < 0 || yy >= WID) continue;
      for (int kx = 0; kx < 7; kx++) {
        const int xx = x + kx - 3;
        if (xx < 0 || xx >= WID) continue;
        a += fb[c * HW + yy * WID + xx] * w[c * 49 + ky * 7 + kx];
      }
    }
    a = a * w[195 + c] + w[198 + c];
    d[c] = a > 0.0f ? a : 0.0f;
  }
  half_t* o = t1 + (size_t)ft * 16 * HW + s;
#pragma unroll
  for (int k = 0; k < 16; k++) {
    float a = w[147 + k * 3] * d[0] + w[147 + k * 3 + 1] * d[1] +
              w[147 + k * 3 + 2] * d[2];
    a = a * w[201 + k] + w[217 + k];
    o[(size_t)k * HW] = (half_t)(a > 0.0f ? a : 0.0f);
  }
}

// ---------------------------------------------------------------------------
// k4/k5: dw 3x3 grouped + BN/ReLU + pw CIN->COUT + BN/ReLU (NCHW f16).
// ---------------------------------------------------------------------------
template <int CIN, int COUT>
__global__ __launch_bounds__(256) void k_dwpw(
    const half_t* __restrict__ in,
    const float* __restrict__ dw, const float* __restrict__ pw,
    const float* __restrict__ sa, const float* __restrict__ ba,
    const float* __restrict__ sb, const float* __restrict__ bb,
    half_t* __restrict__ out)
{
  __shared__ float wdw[CIN * 9];
  __shared__ float wpw[COUT * CIN];
  __shared__ float wsa[CIN], wba[CIN], wsb[COUT], wbb[COUT];
  const int tid = threadIdx.x;
  for (int i = tid; i < CIN * 9; i += 256) wdw[i] = dw[i];
  for (int i = tid; i < COUT * CIN; i += 256) wpw[i] = pw[i];
  for (int i = tid; i < CIN; i += 256) { wsa[i] = sa[i]; wba[i] = ba[i]; }
  for (int i = tid; i < COUT; i += 256) { wsb[i] = sb[i]; wbb[i] = bb[i]; }
  __syncthreads();

  const int gid = blockIdx.x * 256 + tid;
  if (gid >= BT * HW) return;
  const int ft = gid / HW, s = gid % HW;
  const int y = s / WID, x = s % WID;
  const half_t* ip = in + (size_t)ft * CIN * HW;

  float d[CIN];
#pragma unroll
  for (int c = 0; c < CIN; c++) {
    float a = 0.0f;
#pragma unroll
    for (int ky = 0; ky < 3; ky++) {
      const int yy = y + ky - 1;
      if (yy < 0 || yy >= WID) continue;
#pragma unroll
      for (int kx = 0; kx < 3; kx++) {
        const int xx = x + kx - 1;
        if (xx < 0 || xx >= WID) continue;
        a += (float)ip[c * HW + yy * WID + xx] * wdw[c * 9 + ky * 3 + kx];
      }
    }
    a = a * wsa[c] + wba[c];
    d[c] = a > 0.0f ? a : 0.0f;
  }
  half_t* op = out + (size_t)ft * COUT * HW + s;
  for (int k = 0; k < COUT; k++) {
    float a = 0.0f;
#pragma unroll
    for (int c = 0; c < CIN; c++) a += wpw[k * CIN + c] * d[c];
    a = a * wsb[k] + wbb[k];
    op[(size_t)k * HW] = (half_t)(a > 0.0f ? a : 0.0f);
  }
}

// ---------------------------------------------------------------------------
// k6a: dw4 3x3 g64 + BN/ReLU -> NHWC f16, 8-channel vector (b128) stores.
// ---------------------------------------------------------------------------
__global__ __launch_bounds__(256) void k6a_dw4(
    const half_t* __restrict__ t3, const float* __restrict__ dw4,
    const float* __restrict__ s4a, const float* __restrict__ b4a,
    half_t* __restrict__ d4)
{
  __shared__ float wdw[576], wsa[64], wba[64];
  const int tid = threadIdx.x;
  for (int i = tid; i < 576; i += 256) wdw[i] = dw4[i];
  for (int i = tid; i < 64; i += 256) { wsa[i] = s4a[i]; wba[i] = b4a[i]; }
  __syncthreads();

  const int gid = blockIdx.x * 256 + tid;
  if (gid >= BT * HW) return;
  const int ft = gid / HW, s = gid % HW;
  const int y = s / WID, x = s % WID;
  const half_t* ip = t3 + (size_t)ft * 64 * HW;
  half_t* op = d4 + ((size_t)ft * HW + s) * 64;

  for (int c8 = 0; c8 < 8; c8++) {
    v8h h;
#pragma unroll
    for (int j = 0; j < 8; j++) {
      const int c = c8 * 8 + j;
      float a = 0.0f;
#pragma unroll
      for (int ky = 0; ky < 3; ky++) {
        const int yy = y + ky - 1;
        if (yy < 0 || yy >= WID) continue;
#pragma unroll
        for (int kx = 0; kx < 3; kx++) {
          const int xx = x + kx - 1;
          if (xx < 0 || xx >= WID) continue;
          a += (float)ip[c * HW + yy * WID + xx] * wdw[c * 9 + ky * 3 + kx];
        }
      }
      a = a * wsa[c] + wba[c];
      h[j] = (half_t)(a > 0.0f ? a : 0.0f);
    }
    *(v8h*)(op + c8 * 8) = h;
  }
}

// ---------------------------------------------------------------------------
// k6b: out = relu(bn(W4[128,64] @ d4)) + x. Block = 256 threads (8 waves),
// weights loaded once, 7 N-tiles of 16 pixels per block.
// ---------------------------------------------------------------------------
__global__ __launch_bounds__(256) void k6b_pw4_wmma(
    const half_t* __restrict__ d4, const float* __restrict__ pw4,
    const float* __restrict__ s4b, const float* __restrict__ b4b,
    const float* __restrict__ x, float* __restrict__ out)
{
  __shared__ half_t ldsW[128 * 64];   // [row][k]
  __shared__ half_t ldsX[16 * 72];    // [pix][k], pad 8 halves
  __shared__ float  lds_s[128], lds_b[128];

  const int tid = threadIdx.x;
  const int bt  = blockIdx.x / 7;
  const int tg  = blockIdx.x % 7;

  for (int i = tid; i < 128 * 64; i += 256) ldsW[i] = (half_t)pw4[i];
  if (tid < 128) { lds_s[tid] = s4b[tid]; lds_b[tid] = b4b[tid]; }

  const int wave = tid >> 5, lane = tid & 31;
  const int row = lane & 15, kh = lane >> 4;
  const int mbase = wave * 16 + kh * 8;

  for (int t = 0; t < 7; t++) {
    const int n0 = tg * 112 + t * 16;
    for (int i = tid; i < 16 * 64; i += 256) {
      const int px = i >> 6, ch = i & 63;
      ldsX[px * 72 + ch] = d4[((size_t)bt * HW + n0 + px) * 64 + ch];
    }
    __syncthreads();

    v8f acc = {};
#pragma unroll
    for (int kb = 0; kb < 64; kb += 32) {
      v16h a = ld8x2(&ldsW[(wave * 16 + row) * 64 + kb + kh * 8],
                     &ldsW[(wave * 16 + row) * 64 + kb + 16 + kh * 8]);
      v16h bf = ld8x2(&ldsX[row * 72 + kb + kh * 8], &ldsX[row * 72 + kb + 16 + kh * 8]);
      acc = __builtin_amdgcn_wmma_f32_16x16x32_f16(false, a, false, bf,
                                                   (short)0, acc, false, false);
    }

    const int n = n0 + row;
#pragma unroll
    for (int v = 0; v < 8; v++) {
      const int ch = mbase + v;
      float tv = acc[v] * lds_s[ch] + lds_b[ch];
      tv = tv > 0.0f ? tv : 0.0f;
      const size_t idx = ((size_t)bt * 128 + ch) * HW + n;
      out[idx] = tv + x[idx];
    }
    __syncthreads();                   // protect ldsX for next tile
  }
}

// ---------------------------------------------------------------------------
// Launch
// ---------------------------------------------------------------------------
extern "C" void kernel_launch(void* const* d_in, const int* in_sizes, int n_in,
                              void* d_out, int out_size, void* d_ws, size_t ws_size,
                              hipStream_t stream) {
  (void)in_sizes; (void)n_in; (void)out_size; (void)ws_size;
  const float* x     = (const float*)d_in[0];
  const float* w_red = (const float*)d_in[1];
  const float* s_red = (const float*)d_in[2];
  const float* b_red = (const float*)d_in[3];
  const float* dw1   = (const float*)d_in[4];
  const float* pw1   = (const float*)d_in[5];
  const float* s1a   = (const float*)d_in[6];
  const float* b1a   = (const float*)d_in[7];
  const float* s1b   = (const float*)d_in[8];
  const float* b1b   = (const float*)d_in[9];
  const float* dw2   = (const float*)d_in[10];
  const float* pw2   = (const float*)d_in[11];
  const float* s2a   = (const float*)d_in[12];
  const float* b2a   = (const float*)d_in[13];
  const float* s2b   = (const float*)d_in[14];
  const float* b2b   = (const float*)d_in[15];
  const float* dw3   = (const float*)d_in[16];
  const float* pw3   = (const float*)d_in[17];
  const float* s3a   = (const float*)d_in[18];
  const float* b3a   = (const float*)d_in[19];
  const float* s3b   = (const float*)d_in[20];
  const float* b3b   = (const float*)d_in[21];
  const float* dw4   = (const float*)d_in[22];
  const float* pw4   = (const float*)d_in[23];
  const float* s4a   = (const float*)d_in[24];
  const float* b4a   = (const float*)d_in[25];
  const float* s4b   = (const float*)d_in[26];
  const float* b4b   = (const float*)d_in[27];
  const int*   temp  = (const int*)d_in[28];
  float* out = (float*)d_out;

  // Workspace layout (all 256B-aligned offsets), total ~49.2 MB:
  char* ws = (char*)d_ws;
  half_t* feat = (half_t*)(ws);                 // 128*784*64 f16 = 12,845,056
  float*  fbuf = (float*) (ws + 12845056);      // 112*3*784 f32  =  1,053,696
  half_t* t1   = (half_t*)(ws + 13898752);      // 128*16*784 f16 =  3,211,264
  half_t* t2   = (half_t*)(ws + 17110016);      // 128*32*784 f16 =  6,422,528
  half_t* t3   = (half_t*)(ws + 23532544);      // 128*64*784 f16 = 12,845,056
  half_t* d4   = (half_t*)(ws + 36377600);      // 128*784*64 f16 = 12,845,056

  k1_reduce_norm<<<128 * 7, 128, 0, stream>>>(x, w_red, s_red, b_red, feat);
  k2_corr_flow<<<(112 * HW) / K2T, K2T, 0, stream>>>(feat, temp, fbuf);
  k3_dw1_pw1<<<(BT * HW + 255) / 256, 256, 0, stream>>>(fbuf, dw1, pw1,
                                                        s1a, b1a, s1b, b1b, t1);
  k_dwpw<16, 32><<<(BT * HW + 255) / 256, 256, 0, stream>>>(t1, dw2, pw2,
                                                            s2a, b2a, s2b, b2b, t2);
  k_dwpw<32, 64><<<(BT * HW + 255) / 256, 256, 0, stream>>>(t2, dw3, pw3,
                                                            s3a, b3a, s3b, b3b, t3);
  k6a_dw4<<<(BT * HW + 255) / 256, 256, 0, stream>>>(t3, dw4, s4a, b4a, d4);
  k6b_pw4_wmma<<<128 * 7, 256, 0, stream>>>(d4, pw4, s4b, b4b, x, out);
}
